// Single_Encoder_45122926412433
// MI455X (gfx1250) — matrix operand order repaired and checked by reading, compiled, not run on hardware
//
#include <hip/hip_runtime.h>
#include <hip/hip_bf16.h>

typedef float v2f  __attribute__((ext_vector_type(2)));
typedef float v8f  __attribute__((ext_vector_type(8)));
typedef __bf16 v16bf __attribute__((ext_vector_type(16)));

// Problem constants (match reference setup_inputs)
#define NN   20000
#define FF   2000
#define HH   128
#define LL   15
#define NHOP 512          // H*(K+1)
#define KPAD 2048         // FF padded to multiple of 32 (W side)
#define KB16 (KPAD / 16)  // 128 K-blocks of 16

// ---------------------------------------------------------------------------
// Split W1 (8000x128, hop-major rows) into blocked bf16 hi/lo:
//   Wh/Wl[((kb*512) + n)*16 + i]  where k = kb*16+i, n = hop*128+c,
//   source = W1[(hop*2000 + k)*128 + c], zero-padded for k >= 2000.
// Layout: each lane's 16 consecutive K-values for its column are contiguous
// (one aligned 32B load per B fragment).
// ---------------------------------------------------------------------------
__global__ __launch_bounds__(256) void k_repack_w1_bf(const float* __restrict__ W1,
                                                      __bf16* __restrict__ Wh,
                                                      __bf16* __restrict__ Wl) {
    int i = blockIdx.x * 256 + threadIdx.x;       // over 128*512*16 = 1,048,576
    if (i >= KB16 * NHOP * 16) return;
    int ii = i & 15;
    int n  = (i >> 4) & 511;
    int kb = i >> 13;                             // /(16*512)
    int k  = kb * 16 + ii;
    float w = 0.0f;
    if (k < FF) w = W1[((long)(n >> 7) * FF + k) * HH + (n & 127)];
    __bf16 h = (__bf16)w;
    Wh[i] = h;
    Wl[i] = (__bf16)(w - (float)h);
}

// Pad W2 (512x15) -> W2p (512x16), col 15 zeroed
__global__ __launch_bounds__(256) void k_repack_w2(const float* __restrict__ W2,
                                                   float* __restrict__ W2p) {
    int i = blockIdx.x * 256 + threadIdx.x;       // over 512*16
    if (i >= NHOP * 16) return;
    int k = i >> 4;
    int c = i & 15;
    W2p[i] = (c < LL) ? W2[k * LL + c] : 0.0f;
}

// ---------------------------------------------------------------------------
// Convert 8 f32 -> bf16 hi/lo into fragment slots [base..base+7]
// ---------------------------------------------------------------------------
__device__ __forceinline__ void cvt8(float4 q0, float4 q1, int base,
                                     v16bf& hi, v16bf& lo) {
    float f[8] = {q0.x, q0.y, q0.z, q0.w, q1.x, q1.y, q1.z, q1.w};
#pragma unroll
    for (int i = 0; i < 8; ++i) {
        __bf16 h = (__bf16)f[i];
        hi[base + i] = h;
        lo[base + i] = (__bf16)(f[i] - (float)h);
    }
}

__device__ __forceinline__ v8f wmma_bf(v16bf a, v16bf b, v8f c) {
    return __builtin_amdgcn_wmma_f32_16x16x32_bf16(false, a, false, b,
                                                   (short)0, c, false, false);
}

// ---------------------------------------------------------------------------
// GEMM1: U[20000x512] = X[20000x2000] @ W1 (split bf16 hi/lo, fp32 accum)
// acc += Ah*Bh + Al*Bh + Ah*Bl  (3 wmmas per 16x16 tile per K=32)
// Wave tile 32(M) x 64(N); block = 8 waves covers all 512 cols; grid 625.
//
// Fragment layouts used here (16-bit, wave32; A/B in contiguous K order):
//  A 16x32: lane l (m=l&15, h=l>>4): slots 0..15 = K h*16+0..15
//           -> one contiguous 16-float chunk of the row.
//  B 32x16: lane l (n=l&15, h=l>>4): 16 consecutive K = k+16h .. +15 for col n.
//  C/D 16x16: VGPR j -> m = j + 8*(l>>4), n = l&15.
// ---------------------------------------------------------------------------
__global__ __launch_bounds__(256) void k_gemm1_bf16(const float* __restrict__ X,
                                                    const __bf16* __restrict__ Bh,
                                                    const __bf16* __restrict__ Bl,
                                                    float* __restrict__ U) {
    const int lane  = threadIdx.x & 31;
    const int wave  = threadIdx.x >> 5;
    const int mBase = blockIdx.x * 32;
    const int nBase = wave * 64;
    const int lm    = lane & 15;
    const int half  = lane >> 4;

    const float* a0p = X + (long)(mBase + lm) * FF + half * 16;       // rows 0..15
    const float* a1p = X + (long)(mBase + 16 + lm) * FF + half * 16;  // rows 16..31
    const __bf16* bhp = Bh + ((long)half * NHOP + nBase + lm) * 16;
    const __bf16* blp = Bl + ((long)half * NHOP + nBase + lm) * 16;

    v8f acc[2][4];
#pragma unroll
    for (int r = 0; r < 2; ++r)
#pragma unroll
        for (int c = 0; c < 4; ++c)
            acc[r][c] = (v8f){0.f,0.f,0.f,0.f,0.f,0.f,0.f,0.f};

    // 62 full K=32 steps (K 0..1983)
    for (int k = 0; k < 1984; k += 32) {
        v16bf ah0, al0, ah1, al1;
        cvt8(*(const float4*)(a0p),      *(const float4*)(a0p + 4),  0, ah0, al0);
        cvt8(*(const float4*)(a0p + 8),  *(const float4*)(a0p + 12), 8, ah0, al0);
        cvt8(*(const float4*)(a1p),      *(const float4*)(a1p + 4),  0, ah1, al1);
        cvt8(*(const float4*)(a1p + 8),  *(const float4*)(a1p + 12), 8, ah1, al1);
#pragma unroll
        for (int cf = 0; cf < 4; ++cf) {
            v16bf bh = *(const v16bf*)(bhp + cf * 256);
            v16bf bl = *(const v16bf*)(blp + cf * 256);
            acc[0][cf] = wmma_bf(ah0, bh, acc[0][cf]);
            acc[0][cf] = wmma_bf(al0, bh, acc[0][cf]);
            acc[0][cf] = wmma_bf(ah0, bl, acc[0][cf]);
            acc[1][cf] = wmma_bf(ah1, bh, acc[1][cf]);
            acc[1][cf] = wmma_bf(al1, bh, acc[1][cf]);
            acc[1][cf] = wmma_bf(ah1, bl, acc[1][cf]);
        }
        a0p += 32; a1p += 32;
        bhp += 2 * NHOP * 16; blp += 2 * NHOP * 16;
    }

    // Tail step: K 1984..2015; upper half-slots (K 2000..2015) are zero.
    {
        v16bf ah0 = {}, al0 = {}, ah1 = {}, al1 = {};
        if (half == 0) {
            cvt8(*(const float4*)(a0p), *(const float4*)(a0p + 4), 0, ah0, al0);
            cvt8(*(const float4*)(a0p + 8), *(const float4*)(a0p + 12), 8, ah0, al0);
            cvt8(*(const float4*)(a1p), *(const float4*)(a1p + 4), 0, ah1, al1);
            cvt8(*(const float4*)(a1p + 8), *(const float4*)(a1p + 12), 8, ah1, al1);
        }
#pragma unroll
        for (int cf = 0; cf < 4; ++cf) {
            v16bf bh = *(const v16bf*)(bhp + cf * 256);   // zero-padded region
            v16bf bl = *(const v16bf*)(blp + cf * 256);
            acc[0][cf] = wmma_bf(ah0, bh, acc[0][cf]);
            acc[0][cf] = wmma_bf(al0, bh, acc[0][cf]);
            acc[0][cf] = wmma_bf(ah0, bl, acc[0][cf]);
            acc[1][cf] = wmma_bf(ah1, bh, acc[1][cf]);
            acc[1][cf] = wmma_bf(al1, bh, acc[1][cf]);
            acc[1][cf] = wmma_bf(ah1, bl, acc[1][cf]);
        }
    }

    // Store: D layout m = j + 8*half (+16 for second row-frag), n = lm
    const int row0 = mBase + (half << 3);
#pragma unroll
    for (int rf = 0; rf < 2; ++rf)
#pragma unroll
        for (int cf = 0; cf < 4; ++cf)
#pragma unroll
            for (int j = 0; j < 8; ++j)
                U[(long)(row0 + rf * 16 + j) * NHOP + nBase + cf * 16 + lm]
                    = acc[rf][cf][j];
}

// ---------------------------------------------------------------------------
// Copy a 128-wide column slice (row stride ldin floats) into dense Nx128
// ---------------------------------------------------------------------------
__global__ __launch_bounds__(256) void k_copy128(const float* __restrict__ src, int ldin,
                                                 float* __restrict__ dst) {
    int i = blockIdx.x * 256 + threadIdx.x;       // over N*128
    if (i >= NN * HH) return;
    int r = i >> 7, c = i & 127;
    dst[i] = src[(long)r * ldin + c];
}

// ---------------------------------------------------------------------------
// Scatter-add propagation, width 128: tout[dst] += w * tin[src]
// One wave per edge; lane owns 4 contiguous floats (float4 gather from
// L2-resident source); scatter with global_atomic_add_f32.
// ---------------------------------------------------------------------------
__global__ __launch_bounds__(256) void k_prop(const float* __restrict__ tin, int ldin,
                                              float* __restrict__ tout, int ldout,
                                              const int* __restrict__ esrc,
                                              const int* __restrict__ edst,
                                              const float* __restrict__ ew, int E) {
    const int lane = threadIdx.x & 31;
    const int e = blockIdx.x * 8 + (threadIdx.x >> 5);
    if (e >= E) return;
    const int s = esrc[e];
    const int d = edst[e];
    const float w = ew[e];
    const float4 v = *(const float4*)(tin + (long)s * ldin + lane * 4);
    float* po = tout + (long)d * ldout + lane * 4;
    atomicAdd(po + 0, w * v.x);
    atomicAdd(po + 1, w * v.y);
    atomicAdd(po + 2, w * v.z);
    atomicAdd(po + 3, w * v.w);
}

// ---------------------------------------------------------------------------
// h = relu(BN(t + b1)); write into G[:,0:128], zero G[:,128:512]
// ---------------------------------------------------------------------------
__global__ __launch_bounds__(256) void k_bn_relu_pack(const float* __restrict__ t,
                                                      const float* __restrict__ b1,
                                                      const float* __restrict__ gamma,
                                                      const float* __restrict__ beta,
                                                      const float* __restrict__ mean,
                                                      const float* __restrict__ var,
                                                      float* __restrict__ G) {
    int i = blockIdx.x * 256 + threadIdx.x;       // over N*512
    if (i >= NN * NHOP) return;
    int c = i & 511;
    float v = 0.0f;
    if (c < HH) {
        int r = i >> 9;
        float u = t[(long)r * HH + c] + b1[c];
        u = (u - mean[c]) * rsqrtf(var[c] + 1e-3f) * gamma[c] + beta[c];
        v = fmaxf(u, 0.0f);
    }
    G[i] = v;
}

// ---------------------------------------------------------------------------
// GEMM2: Z[20000x15] = G[20000x512] @ W2p[512x16] + b2   (col 15 discarded)
// One wave per 16-row block; fp32 WMMA 16x16x4, K=512 (trivial FLOPs).
// ---------------------------------------------------------------------------
__global__ __launch_bounds__(256) void k_gemm2_wmma(const float* __restrict__ G,
                                                    const float* __restrict__ B,
                                                    const float* __restrict__ bias,
                                                    float* __restrict__ Z) {
    const int lane = threadIdx.x & 31;
    const int blk = blockIdx.x * 8 + (threadIdx.x >> 5);
    if (blk >= NN / 16) return;                    // uniform per wave
    const int mBase = blk * 16;
    const int lm = lane & 15;
    const int kb = (lane >> 4) << 1;

    const float* ap = G + (long)(mBase + lm) * NHOP + kb;
    const float* bp = B + kb * 16 + lm;

    v8f acc = {0.f,0.f,0.f,0.f,0.f,0.f,0.f,0.f};
    for (int k = 0; k < NHOP; k += 4) {
        v2f a = *(const v2f*)ap;
        v2f b; b.x = bp[0]; b.y = bp[16];
        acc = __builtin_amdgcn_wmma_f32_16x16x4_f32(false, a, false, b, (short)0, acc, false, false);
        ap += 4; bp += 64;
    }

    if (lm < LL) {
        const int row0 = mBase + ((lane >> 4) << 3);
        const float bb = bias[lm];
#pragma unroll
        for (int j = 0; j < 8; ++j)
            Z[(long)(row0 + j) * LL + lm] = acc[j] + bb;
    }
}

// ---------------------------------------------------------------------------
extern "C" void kernel_launch(void* const* d_in, const int* in_sizes, int n_in,
                              void* d_out, int out_size, void* d_ws, size_t ws_size,
                              hipStream_t stream) {
    const float* x     = (const float*)d_in[0];   // 20000 x 2000
    const int*   esrc  = (const int*)  d_in[1];   // E
    const int*   edst  = (const int*)  d_in[2];   // E
    const float* ew    = (const float*)d_in[3];   // E
    const float* W1    = (const float*)d_in[4];   // 8000 x 128
    const float* b1    = (const float*)d_in[5];   // 128
    const float* gamma = (const float*)d_in[6];   // 128
    const float* beta  = (const float*)d_in[7];   // 128
    const float* mmean = (const float*)d_in[8];   // 128
    const float* mvar  = (const float*)d_in[9];   // 128
    const float* W2    = (const float*)d_in[10];  // 512 x 15
    const float* b2    = (const float*)d_in[11];  // 15
    float* out = (float*)d_out;                   // 20000 x 15
    const int E = in_sizes[1];

    // Workspace layout
    const long nWbf = (long)KB16 * NHOP * 16;     // 1,048,576 bf16 per buffer
    __bf16* W1bh = (__bf16*)d_ws;
    __bf16* W1bl = W1bh + nWbf;
    float*  W2p  = (float*)(W1bl + nWbf);         // 512*16 f32 (offset 4MB, aligned)
    float*  U    = W2p + (long)NHOP * 16;         // 20000*512 f32
    float*  Oa   = U   + (long)NN * NHOP;         // 20000*128 f32
    float*  Ob   = Oa  + (long)NN * HH;           // 20000*128 f32
    float*  G    = U;                             // reuse U after layer-1 Horner

    // 1) Repack/split weights
    k_repack_w1_bf<<<(int)((nWbf + 255) / 256), 256, 0, stream>>>(W1, W1bh, W1bl);
    k_repack_w2<<<(NHOP * 16 + 255) / 256, 256, 0, stream>>>(W2, W2p);

    // 2) U = x @ W1r  (split-bf16 WMMA, fp32 accumulate)
    k_gemm1_bf16<<<NN / 32, 256, 0, stream>>>(x, W1bh, W1bl, U);

    // 3) Layer-1 Horner at width 128: h_pre = U0 + A(U1 + A(U2 + A*U3))
    const int nCopyBlk = (NN * HH + 255) / 256;
    const int nPropBlk = (E + 7) / 8;
    k_copy128<<<nCopyBlk, 256, 0, stream>>>(U + 2 * HH, NHOP, Oa);           // Oa = U2
    k_prop<<<nPropBlk, 256, 0, stream>>>(U + 3 * HH, NHOP, Oa, HH, esrc, edst, ew, E);
    k_copy128<<<nCopyBlk, 256, 0, stream>>>(U + 1 * HH, NHOP, Ob);           // Ob = U1
    k_prop<<<nPropBlk, 256, 0, stream>>>(Oa, HH, Ob, HH, esrc, edst, ew, E);
    k_copy128<<<nCopyBlk, 256, 0, stream>>>(U + 0 * HH, NHOP, Oa);           // Oa = U0
    k_prop<<<nPropBlk, 256, 0, stream>>>(Ob, HH, Oa, HH, esrc, edst, ew, E); // Oa = h_pre

    // 4) BN + ReLU -> G[:,0:128], zero remaining hop slots
    k_bn_relu_pack<<<(NN * NHOP + 255) / 256, 256, 0, stream>>>(
        Oa, b1, gamma, beta, mmean, mvar, G);

    // 5) Layer-2 hops at width 128 into G column blocks
    k_prop<<<nPropBlk, 256, 0, stream>>>(G + 0 * HH, NHOP, G + 1 * HH, NHOP, esrc, edst, ew, E);
    k_prop<<<nPropBlk, 256, 0, stream>>>(G + 1 * HH, NHOP, G + 2 * HH, NHOP, esrc, edst, ew, E);
    k_prop<<<nPropBlk, 256, 0, stream>>>(G + 2 * HH, NHOP, G + 3 * HH, NHOP, esrc, edst, ew, E);

    // 6) Z = G @ W2 + b2
    k_gemm2_wmma<<<(NN / 16 + 7) / 8, 256, 0, stream>>>(G, W2p, b2, out);
}
